// TokenSubNet_27462020891243
// MI455X (gfx1250) — matrix operand order repaired and checked by reading, not compiled, not run on hardware
//
#include <hip/hip_runtime.h>
#include <hip/hip_bf16.h>
#include <math.h>

#define EMBED 192
#define DH    384
#define DTRK  12
#define DSTN  16
#define KDIR  4
#define BN    2
#define MAXM  8192

typedef __bf16 bf16;
typedef __attribute__((ext_vector_type(16))) __bf16 v16bf;
typedef __attribute__((ext_vector_type(8)))  float  v8f;
typedef __attribute__((ext_vector_type(4)))  unsigned int u32x4;

union Frag { v16bf v; u32x4 q[2]; };

__device__ __forceinline__ float gelu_exact(float x) {
  return 0.5f * x * (1.0f + erff(x * 0.70710678118654752f));
}
__device__ __forceinline__ float silu_f(float x) {
  return x / (1.0f + __expf(-x));
}

// ---------------------------------------------------------------------------
// GEMM: Y[M,N] = A[M,K](bf16) * W[N,K]^T(bf16)  (+bias)(+gelu)(+resid)
// wave tile 16x(NT*16), block = 4 waves (64 rows). NT is compile-time so the
// inner loop has zero control flow: no accumulator merges / v_mov storms.
// M multiple of 64; K multiple of 32.
// ---------------------------------------------------------------------------
template <int NT>
__global__ void __launch_bounds__(128)
k_gemm_t(const bf16* __restrict__ A, const bf16* __restrict__ Wm,
         const float* __restrict__ bias, const float* __restrict__ resid,
         float* __restrict__ Cf, bf16* __restrict__ Cb,
         int M, int N, int K, int act)
{
  const int wave = threadIdx.x >> 5;
  const int lane = threadIdx.x & 31;
  const int h    = lane >> 4;      // lane half
  const int r16  = lane & 15;
  const int m0   = blockIdx.y * 64 + wave * 16;
  const int nb   = blockIdx.x * (NT * 16);

  v8f acc[NT];
#pragma unroll
  for (int j = 0; j < NT; ++j) acc[j] = (v8f)0.0f;

  const bf16* arow = A + (size_t)(m0 + r16) * K + h * 8;
  // per-tile B row base: column n = nb + j*16 + r16, k offset h*8 (elems 0..7 -> h*8.., 8..15 -> 16+h*8..)
  const bf16* wrow = Wm + (size_t)(nb + r16) * K + h * 8;

  for (int k0 = 0; k0 < K; k0 += 32) {
    Frag a;
    // A 16x32 fragment: elems 0..7 -> k0+h*8.., elems 8..15 -> k0+16+h*8..
    a.q[0] = *(const u32x4*)(arow + k0);
    a.q[1] = *(const u32x4*)(arow + k0 + 16);
    if (k0 + 32 < K) __builtin_prefetch(arow + k0 + 32, 0, 1);
#pragma unroll
    for (int j = 0; j < NT; ++j) {
      const bf16* wr = wrow + (size_t)j * 16 * K + k0;
      Frag b;
      b.q[0] = *(const u32x4*)(wr);
      b.q[1] = *(const u32x4*)(wr + 16);
      acc[j] = __builtin_amdgcn_wmma_f32_16x16x32_bf16(
          false, a.v, false, b.v, (short)0, acc[j], false, false);
    }
  }

#pragma unroll
  for (int j = 0; j < NT; ++j) {
    int gn = nb + j * 16 + r16;
    float bv = bias ? bias[gn] : 0.0f;
#pragma unroll
    for (int e = 0; e < 8; ++e) {
      int gm = m0 + e + 8 * h;
      float v = acc[j][e] + bv;
      if (act == 1) v = gelu_exact(v);
      size_t o = (size_t)gm * N + gn;
      if (resid) v += resid[o];
      if (Cf) Cf[o] = v;
      if (Cb) Cb[o] = (bf16)v;
    }
  }
}

static void gemm(hipStream_t st, const bf16* A, const bf16* Wm, const float* bias,
                 const float* resid, float* Cf, bf16* Cb, int M, int N, int K, int act)
{
  if ((N & 63) == 0)
    k_gemm_t<4><<<dim3(N / 64, M / 64), 128, 0, st>>>(A, Wm, bias, resid, Cf, Cb, M, N, K, act);
  else // N == 48 (x_proj, padded): 3 tiles, one block column
    k_gemm_t<3><<<dim3(1, M / 64), 128, 0, st>>>(A, Wm, bias, resid, Cf, Cb, M, N, K, act);
}

// ---------------------------------------------------------------------------
// LayerNorm over last dim C (one wave per token), f32 + optional bf16 out
// ---------------------------------------------------------------------------
__global__ void __launch_bounds__(128)
k_layernorm(const float* __restrict__ x, const float* __restrict__ g,
            const float* __restrict__ b, float* of, bf16* ob, int M, int C)
{
  int token = blockIdx.x * 4 + (threadIdx.x >> 5);
  int lane = threadIdx.x & 31;
  if (token >= M) return;
  const float* xr = x + (size_t)token * C;
  float s = 0.f, s2 = 0.f;
  for (int c = lane; c < C; c += 32) { float v = xr[c]; s += v; s2 += v * v; }
  for (int o = 16; o > 0; o >>= 1) { s += __shfl_xor(s, o, 32); s2 += __shfl_xor(s2, o, 32); }
  float mean = s / C;
  float rstd = rsqrtf(s2 / C - mean * mean + 1e-5f);
  for (int c = lane; c < C; c += 32) {
    float v = (xr[c] - mean) * rstd * g[c] + b[c];
    size_t o = (size_t)token * C + c;
    if (of) of[o] = v;
    if (ob) ob[o] = (bf16)v;
  }
}

// out-norm over DH + gate with silu(z) -> bf16
__global__ void __launch_bounds__(128)
k_gate(const float* __restrict__ y, const float* __restrict__ xz,
       const float* __restrict__ g, const float* __restrict__ b,
       bf16* __restrict__ out, int M)
{
  int token = blockIdx.x * 4 + (threadIdx.x >> 5);
  int lane = threadIdx.x & 31;
  if (token >= M) return;
  const float* yr = y + (size_t)token * DH;
  float s = 0.f, s2 = 0.f;
  for (int c = lane; c < DH; c += 32) { float v = yr[c]; s += v; s2 += v * v; }
  for (int o = 16; o > 0; o >>= 1) { s += __shfl_xor(s, o, 32); s2 += __shfl_xor(s2, o, 32); }
  float mean = s / DH;
  float rstd = rsqrtf(s2 / DH - mean * mean + 1e-5f);
  const float* zr = xz + (size_t)token * (2 * DH) + DH;
  for (int c = lane; c < DH; c += 32) {
    float v = (yr[c] - mean) * rstd * g[c] + b[c];
    out[(size_t)token * DH + c] = (bf16)(v * silu_f(zr[c]));
  }
}

// ---------------------------------------------------------------------------
// depthwise 3x3 conv + SiLU on xh half of xz; emit HW and WH token orders
// ---------------------------------------------------------------------------
__global__ void __launch_bounds__(DH)
k_conv(const float* __restrict__ xz, const float* __restrict__ cw,
       const float* __restrict__ cb, float* __restrict__ fhw,
       float* __restrict__ fwh, bf16* __restrict__ bhw, bf16* __restrict__ bwh,
       int H, int W)
{
  int d = threadIdx.x;
  int l = blockIdx.x;
  int b = blockIdx.y;
  int L = H * W;
  int hh = l / W, ww = l % W;
  float acc = cb[d];
#pragma unroll
  for (int dy = -1; dy <= 1; ++dy)
#pragma unroll
    for (int dx = -1; dx <= 1; ++dx) {
      int yy = hh + dy, xx = ww + dx;
      if (yy >= 0 && yy < H && xx >= 0 && xx < W)
        acc += xz[((size_t)(b * L + yy * W + xx)) * (2 * DH) + d] *
               cw[d * 9 + (dy + 1) * 3 + (dx + 1)];
    }
  float v = silu_f(acc);
  size_t ohw = ((size_t)(b * L + l)) * DH + d;
  size_t owh = ((size_t)(b * L + ww * H + hh)) * DH + d;
  fhw[ohw] = v; bhw[ohw] = (bf16)v;
  fwh[owh] = v; bwh[owh] = (bf16)v;
}

// extract dt-rank rows (cols 0..11 of xdbl) into padded (M,32) bf16 per direction
__global__ void k_dtr(const float* __restrict__ bc, bf16* __restrict__ dtr, int Mtok)
{
  int idx = blockIdx.x * 256 + threadIdx.x;
  int total = KDIR * Mtok * 32;
  if (idx >= total) return;
  int r = idx % 32;
  int row = (idx / 32) % Mtok;
  int k = idx / (32 * Mtok);
  float v = (r < DTRK) ? bc[((size_t)k * Mtok + row) * 48 + r] : 0.0f;
  dtr[idx] = (bf16)v;
}

// ---------------------------------------------------------------------------
// selective scan: grid (DH/64, B, K), thread = one d channel, 16-state recur.
// dts and ys may alias (in-place): read-then-write per element per thread.
// ---------------------------------------------------------------------------
__global__ void __launch_bounds__(64)
k_scan(const float* __restrict__ x_hw, const float* __restrict__ x_wh,
       const float* dts, const float* __restrict__ bc, float* ys,
       const float* __restrict__ A_log, const float* __restrict__ dt_b,
       const float* __restrict__ Ds, int L)
{
  int d = blockIdx.x * 64 + threadIdx.x;
  int b = blockIdx.y;
  int k = blockIdx.z;
  const float* xk  = (k & 1) ? x_wh : x_hw;
  const float* dk  = dts + (size_t)k * BN * L * DH;
  const float* bck = bc  + (size_t)k * BN * L * 48;
  float*       yk  = ys  + (size_t)k * BN * L * DH;
  float A[DSTN];
#pragma unroll
  for (int n = 0; n < DSTN; ++n)
    A[n] = -__expf(A_log[((size_t)k * DH + d) * DSTN + n]);
  float dtb = dt_b[k * DH + d];
  float Dd  = Ds[k * DH + d];
  float h[DSTN];
#pragma unroll
  for (int n = 0; n < DSTN; ++n) h[n] = 0.f;
  int rev = (k >= 2);
  for (int s = 0; s < L; ++s) {
    int t = rev ? (L - 1 - s) : s;
    size_t row = (size_t)b * L + t;
    float xv = xk[row * DH + d];
    float dt = dk[row * DH + d] + dtb;
    dt = (dt > 20.f) ? dt : log1pf(__expf(dt));
    const float* bcr = bck + row * 48;
    float dx = dt * xv;
    float y = 0.f;
#pragma unroll
    for (int n = 0; n < DSTN; ++n) {
      h[n] = __expf(dt * A[n]) * h[n] + dx * bcr[12 + n];
      y += h[n] * bcr[28 + n];
    }
    yk[row * DH + d] = y + Dd * xv;
  }
}

// combine the 4 direction outputs (WH dirs read at transposed token index)
__global__ void k_combine(const float* __restrict__ ys, float* __restrict__ out,
                          int H, int W)
{
  int idx = blockIdx.x * 256 + threadIdx.x;
  int L = H * W;
  int total = BN * L * DH;
  if (idx >= total) return;
  int d = idx % DH;
  int l = (idx / DH) % L;
  int b = idx / (DH * L);
  int hh = l / W, ww = l % W;
  int lwh = ww * H + hh;
  size_t stride = (size_t)BN * L * DH;
  size_t rhw = ((size_t)b * L + l) * DH + d;
  size_t rwh = ((size_t)b * L + lwh) * DH + d;
  out[rhw] = ys[rhw] + ys[2 * stride + rhw] + ys[stride + rwh] + ys[3 * stride + rwh];
}

// -------------------- weight conversion --------------------
__global__ void k_cvt(const float* __restrict__ s, bf16* __restrict__ d, int n)
{ int i = blockIdx.x * 256 + threadIdx.x; if (i < n) d[i] = (bf16)s[i]; }

__global__ void k_cvt_pad(const float* __restrict__ s, bf16* __restrict__ d,
                          int rows, int sc, int dc)
{
  int i = blockIdx.x * 256 + threadIdx.x;
  if (i >= rows * dc) return;
  int c = i % dc, r = i / dc;
  d[i] = (bf16)((c < sc) ? s[(size_t)r * sc + c] : 0.0f);
}

__global__ void k_cvt_xproj(const float* __restrict__ s, bf16* __restrict__ d)
{
  int i = blockIdx.x * 256 + threadIdx.x;
  if (i >= KDIR * 48 * DH) return;
  int c = i % DH;
  int r = (i / DH) % 48;
  int k = i / (DH * 48);
  d[i] = (bf16)((r < 44) ? s[((size_t)k * 44 + r) * DH + c] : 0.0f);
}

// -------------------- pyramid ops (channels-last) --------------------
__global__ void k_up(const float* __restrict__ in, float* __restrict__ out,
                     int Hin, int Win)
{
  int idx = blockIdx.x * 256 + threadIdx.x;
  int Ho = Hin * 2, Wo = Win * 2;
  int total = BN * Ho * Wo * EMBED;
  if (idx >= total) return;
  int c  = idx % EMBED;
  int ow = (idx / EMBED) % Wo;
  int oh = (idx / (EMBED * Wo)) % Ho;
  int b  = idx / (EMBED * Wo * Ho);
  float sy = oh * 0.5f - 0.25f, sx = ow * 0.5f - 0.25f;
  int y0 = (int)floorf(sy), x0 = (int)floorf(sx);
  float fy = sy - y0, fx = sx - x0;
  int y1 = y0 + 1, x1 = x0 + 1;
  y0 = y0 < 0 ? 0 : (y0 > Hin - 1 ? Hin - 1 : y0);
  y1 = y1 < 0 ? 0 : (y1 > Hin - 1 ? Hin - 1 : y1);
  x0 = x0 < 0 ? 0 : (x0 > Win - 1 ? Win - 1 : x0);
  x1 = x1 < 0 ? 0 : (x1 > Win - 1 ? Win - 1 : x1);
  const float* base = in + (size_t)b * Hin * Win * EMBED;
  float v00 = base[((size_t)y0 * Win + x0) * EMBED + c];
  float v01 = base[((size_t)y0 * Win + x1) * EMBED + c];
  float v10 = base[((size_t)y1 * Win + x0) * EMBED + c];
  float v11 = base[((size_t)y1 * Win + x1) * EMBED + c];
  out[idx] = (1 - fy) * ((1 - fx) * v00 + fx * v01) + fy * ((1 - fx) * v10 + fx * v11);
}

__global__ void k_down(const float* __restrict__ in, float* __restrict__ out,
                       int Ho, int Wo)
{
  int idx = blockIdx.x * 256 + threadIdx.x;
  int total = BN * Ho * Wo * EMBED;
  if (idx >= total) return;
  int c = idx % EMBED;
  int w = (idx / EMBED) % Wo;
  int h = (idx / (EMBED * Wo)) % Ho;
  int b = idx / (EMBED * Wo * Ho);
  int Win = 2 * Wo;
  const float* base = in + (size_t)b * (2 * Ho) * Win * EMBED;
  float v = base[((size_t)(2 * h) * Win + 2 * w) * EMBED + c] +
            base[((size_t)(2 * h) * Win + 2 * w + 1) * EMBED + c] +
            base[((size_t)(2 * h + 1) * Win + 2 * w) * EMBED + c] +
            base[((size_t)(2 * h + 1) * Win + 2 * w + 1) * EMBED + c];
  out[idx] = 0.25f * v;
}

__global__ void k_add(float* __restrict__ d, const float* __restrict__ s, int n)
{ int i = blockIdx.x * 256 + threadIdx.x; if (i < n) d[i] += s[i]; }

__global__ void k_alpha(const float* __restrict__ tok, const float* __restrict__ ref,
                        const float* __restrict__ alpha, float* __restrict__ out, int n)
{
  int i = blockIdx.x * 256 + threadIdx.x;
  if (i >= n) return;
  int c = i % EMBED;
  float a = alpha[c];
  float sa = (a > 0.f) ? 1.f : (a < 0.f ? -1.f : 0.f);
  out[i] = sa * fmaxf(fabsf(a), 1e-3f) * tok[i] + ref[i];
}

__global__ void k_nchw(const float* __restrict__ in, float* __restrict__ out,
                       int H, int W)
{
  int idx = blockIdx.x * 256 + threadIdx.x;
  int total = BN * H * W * EMBED;
  if (idx >= total) return;
  int c = idx % EMBED;
  int w = (idx / EMBED) % W;
  int h = (idx / (EMBED * W)) % H;
  int b = idx / (EMBED * W * H);
  out[(((size_t)b * EMBED + c) * H + h) * W + w] = in[idx];
}

// ---------------------------------------------------------------------------
// host orchestration
// ---------------------------------------------------------------------------
struct Ws {
  float *F0, *F1, *TMP, *FL[4];
  bf16  *LNB;
  float *XZ, *XCHWF, *XCWHF;
  bf16  *XCHWB, *XCWHB;
  float *BC; bf16 *DTRB; float *DTS; float *YC;
  bf16  *YG, *MLPH;
  bf16  *WI, *WX, *WD, *WO, *WF1, *WF2;
};

struct Par {
  const float *ln1_g, *ln1_b, *in_proj_w, *conv_w, *conv_b, *x_proj_w;
  const float *dt_w, *dt_b, *A_log, *Ds, *onorm_g, *onorm_b, *out_proj_w;
  const float *ln2_g, *ln2_b, *fc1_w, *fc1_b, *fc2_w, *fc2_b;
};

static void run_vss(hipStream_t st, Ws& w, const Par& p, int H, int W)
{
  int L = H * W, M = BN * L;
  // weights -> bf16 (padded where needed)
  k_cvt      <<<(768 * 192 + 255) / 256, 256, 0, st>>>(p.in_proj_w, w.WI, 768 * 192);
  k_cvt_xproj<<<(KDIR * 48 * DH + 255) / 256, 256, 0, st>>>(p.x_proj_w, w.WX);
  k_cvt_pad  <<<(KDIR * DH * 32 + 255) / 256, 256, 0, st>>>(p.dt_w, w.WD, KDIR * DH, DTRK, 32);
  k_cvt      <<<(192 * 384 + 255) / 256, 256, 0, st>>>(p.out_proj_w, w.WO, 192 * 384);
  k_cvt      <<<(768 * 192 + 255) / 256, 256, 0, st>>>(p.fc1_w, w.WF1, 768 * 192);
  k_cvt      <<<(192 * 768 + 255) / 256, 256, 0, st>>>(p.fc2_w, w.WF2, 192 * 768);

  // ln1(x) -> bf16
  k_layernorm<<<(M + 3) / 4, 128, 0, st>>>(w.F0, p.ln1_g, p.ln1_b, nullptr, w.LNB, M, EMBED);
  // in_proj GEMM -> xz (M,768)
  gemm(st, w.LNB, w.WI, nullptr, nullptr, w.XZ, nullptr, M, 768, EMBED, 0);
  // dwconv + silu -> xc (HW & WH orders, f32 + bf16)
  k_conv<<<dim3(L, BN), DH, 0, st>>>(w.XZ, p.conv_w, p.conv_b, w.XCHWF, w.XCWHF,
                                     w.XCHWB, w.XCWHB, H, W);
  // x_proj GEMMs (per direction, N=48 padded)
  for (int k = 0; k < KDIR; ++k) {
    const bf16* A = (k & 1) ? w.XCWHB : w.XCHWB;
    gemm(st, A, w.WX + (size_t)k * 48 * DH, nullptr, nullptr,
         w.BC + (size_t)k * M * 48, nullptr, M, 48, DH, 0);
  }
  // dt-rank extract + dt GEMMs (K padded 12->32)
  k_dtr<<<(KDIR * M * 32 + 255) / 256, 256, 0, st>>>(w.BC, w.DTRB, M);
  for (int k = 0; k < KDIR; ++k)
    gemm(st, w.DTRB + (size_t)k * M * 32, w.WD + (size_t)k * DH * 32, nullptr, nullptr,
         w.DTS + (size_t)k * M * DH, nullptr, M, DH, 32, 0);
  // selective scan (in-place over DTS)
  k_scan<<<dim3(DH / 64, BN, KDIR), 64, 0, st>>>(w.XCHWF, w.XCWHF, w.DTS, w.BC, w.DTS,
                                                 p.A_log, p.dt_b, p.Ds, L);
  // combine 4 directions
  k_combine<<<(M * DH + 255) / 256, 256, 0, st>>>(w.DTS, w.YC, H, W);
  // out-norm + silu(z) gate -> bf16
  k_gate<<<(M + 3) / 4, 128, 0, st>>>(w.YC, w.XZ, p.onorm_g, p.onorm_b, w.YG, M);
  // out_proj GEMM + residual(x) -> F1
  gemm(st, w.YG, w.WO, nullptr, w.F0, w.F1, nullptr, M, EMBED, DH, 0);
  // MLP: ln2 -> fc1(+bias,gelu)->bf16 -> fc2(+bias,+resid) -> F0
  k_layernorm<<<(M + 3) / 4, 128, 0, st>>>(w.F1, p.ln2_g, p.ln2_b, nullptr, w.LNB, M, EMBED);
  gemm(st, w.LNB, w.WF1, p.fc1_b, nullptr, nullptr, w.MLPH, M, 768, EMBED, 1);
  gemm(st, w.MLPH, w.WF2, p.fc2_b, w.F1, w.F0, nullptr, M, EMBED, 768, 0);
}

static void run_refine(hipStream_t st, Ws& w, const Par& base, int r, int H, int W)
{
  for (int dep = 0; dep < 3; ++dep) {
    int i = r * 3 + dep;
    Par p;
    p.ln1_g = base.ln1_g + (size_t)i * EMBED;
    p.ln1_b = base.ln1_b + (size_t)i * EMBED;
    p.in_proj_w = base.in_proj_w + (size_t)i * 768 * EMBED;
    p.conv_w = base.conv_w + (size_t)i * DH * 9;
    p.conv_b = base.conv_b + (size_t)i * DH;
    p.x_proj_w = base.x_proj_w + (size_t)i * KDIR * 44 * DH;
    p.dt_w = base.dt_w + (size_t)i * KDIR * DH * DTRK;
    p.dt_b = base.dt_b + (size_t)i * KDIR * DH;
    p.A_log = base.A_log + (size_t)i * KDIR * DH * DSTN;
    p.Ds = base.Ds + (size_t)i * KDIR * DH;
    p.onorm_g = base.onorm_g + (size_t)i * DH;
    p.onorm_b = base.onorm_b + (size_t)i * DH;
    p.out_proj_w = base.out_proj_w + (size_t)i * EMBED * DH;
    p.ln2_g = base.ln2_g + (size_t)i * EMBED;
    p.ln2_b = base.ln2_b + (size_t)i * EMBED;
    p.fc1_w = base.fc1_w + (size_t)i * 768 * EMBED;
    p.fc1_b = base.fc1_b + (size_t)i * 768;
    p.fc2_w = base.fc2_w + (size_t)i * EMBED * 768;
    p.fc2_b = base.fc2_b + (size_t)i * EMBED;
    run_vss(st, w, p, H, W);
  }
}

extern "C" void kernel_launch(void* const* d_in, const int* in_sizes, int n_in,
                              void* d_out, int out_size, void* d_ws, size_t ws_size,
                              hipStream_t stream)
{
  if (n_in < 27) return;
  const float* tok[4] = {(const float*)d_in[0], (const float*)d_in[1],
                         (const float*)d_in[2], (const float*)d_in[3]};
  Par base;
  base.ln1_g = (const float*)d_in[4];   base.ln1_b = (const float*)d_in[5];
  base.in_proj_w = (const float*)d_in[6];
  base.conv_w = (const float*)d_in[7];  base.conv_b = (const float*)d_in[8];
  base.x_proj_w = (const float*)d_in[9];
  base.dt_w = (const float*)d_in[10];   base.dt_b = (const float*)d_in[11];
  base.A_log = (const float*)d_in[12];  base.Ds = (const float*)d_in[13];
  base.onorm_g = (const float*)d_in[14]; base.onorm_b = (const float*)d_in[15];
  base.out_proj_w = (const float*)d_in[16];
  base.ln2_g = (const float*)d_in[17];  base.ln2_b = (const float*)d_in[18];
  base.fc1_w = (const float*)d_in[19];  base.fc1_b = (const float*)d_in[20];
  base.fc2_w = (const float*)d_in[21];  base.fc2_b = (const float*)d_in[22];
  const float* alpha[4] = {(const float*)d_in[23], (const float*)d_in[24],
                           (const float*)d_in[25], (const float*)d_in[26]};

  // ---- carve workspace ----
  char* wsb = (char*)d_ws;
  size_t off = 0;
  auto alloc = [&](size_t bytes) -> void* {
    bytes = (bytes + 255) & ~(size_t)255;
    void* p = wsb + off; off += bytes; return p;
  };
  Ws w;
  w.F0  = (float*)alloc((size_t)MAXM * EMBED * 4);
  w.F1  = (float*)alloc((size_t)MAXM * EMBED * 4);
  w.TMP = (float*)alloc((size_t)MAXM * EMBED * 4);
  int Hs[4] = {64, 32, 16, 8};
  for (int i = 0; i < 4; ++i)
    w.FL[i] = (float*)alloc((size_t)BN * Hs[i] * Hs[i] * EMBED * 4);
  w.LNB   = (bf16*)alloc((size_t)MAXM * EMBED * 2);
  w.XZ    = (float*)alloc((size_t)MAXM * 768 * 4);
  w.XCHWF = (float*)alloc((size_t)MAXM * DH * 4);
  w.XCWHF = (float*)alloc((size_t)MAXM * DH * 4);
  w.XCHWB = (bf16*)alloc((size_t)MAXM * DH * 2);
  w.XCWHB = (bf16*)alloc((size_t)MAXM * DH * 2);
  w.BC    = (float*)alloc((size_t)KDIR * MAXM * 48 * 4);
  w.DTRB  = (bf16*)alloc((size_t)KDIR * MAXM * 32 * 2);
  w.DTS   = (float*)alloc((size_t)KDIR * MAXM * DH * 4);
  w.YC    = (float*)alloc((size_t)MAXM * DH * 4);
  w.YG    = (bf16*)alloc((size_t)MAXM * DH * 2);
  w.MLPH  = (bf16*)alloc((size_t)MAXM * 768 * 2);
  w.WI  = (bf16*)alloc((size_t)768 * 192 * 2);
  w.WX  = (bf16*)alloc((size_t)KDIR * 48 * DH * 2);
  w.WD  = (bf16*)alloc((size_t)KDIR * DH * 32 * 2);
  w.WO  = (bf16*)alloc((size_t)192 * 384 * 2);
  w.WF1 = (bf16*)alloc((size_t)768 * 192 * 2);
  w.WF2 = (bf16*)alloc((size_t)192 * 768 * 2);

  int n0 = BN * 64 * 64 * EMBED, n1 = BN * 32 * 32 * EMBED;
  int n2 = BN * 16 * 16 * EMBED, n3 = BN * 8 * 8 * EMBED;

  // ---- level 0: f0 = a0*t0 + refine(up(t1)) ----
  k_up<<<(n0 + 255) / 256, 256, 0, stream>>>(tok[1], w.F0, 32, 32);
  run_refine(stream, w, base, 0, 64, 64);
  k_alpha<<<(n0 + 255) / 256, 256, 0, stream>>>(tok[0], w.F0, alpha[0], w.FL[0], n0);

  // ---- level 1: f1 = a1*t1 + refine(up(t2) + down(f0)) ----
  k_up<<<(n1 + 255) / 256, 256, 0, stream>>>(tok[2], w.F0, 16, 16);
  k_down<<<(n1 + 255) / 256, 256, 0, stream>>>(w.FL[0], w.TMP, 32, 32);
  k_add<<<(n1 + 255) / 256, 256, 0, stream>>>(w.F0, w.TMP, n1);
  run_refine(stream, w, base, 1, 32, 32);
  k_alpha<<<(n1 + 255) / 256, 256, 0, stream>>>(tok[1], w.F0, alpha[1], w.FL[1], n1);

  // ---- level 2: f2 = a2*t2 + refine(up(t3) + down(f1)) ----
  k_up<<<(n2 + 255) / 256, 256, 0, stream>>>(tok[3], w.F0, 8, 8);
  k_down<<<(n2 + 255) / 256, 256, 0, stream>>>(w.FL[1], w.TMP, 16, 16);
  k_add<<<(n2 + 255) / 256, 256, 0, stream>>>(w.F0, w.TMP, n2);
  run_refine(stream, w, base, 2, 16, 16);
  k_alpha<<<(n2 + 255) / 256, 256, 0, stream>>>(tok[2], w.F0, alpha[2], w.FL[2], n2);

  // ---- level 3: f3 = a3*t3 + refine(down(f2)) ----
  k_down<<<(n3 + 255) / 256, 256, 0, stream>>>(w.FL[2], w.F0, 8, 8);
  run_refine(stream, w, base, 3, 8, 8);
  k_alpha<<<(n3 + 255) / 256, 256, 0, stream>>>(tok[3], w.F0, alpha[3], w.FL[3], n3);

  // ---- emit NCHW outputs, concatenated f0,f1,f2,f3 ----
  float* out = (float*)d_out;
  k_nchw<<<(n0 + 255) / 256, 256, 0, stream>>>(w.FL[0], out, 64, 64);
  k_nchw<<<(n1 + 255) / 256, 256, 0, stream>>>(w.FL[1], out + n0, 32, 32);
  k_nchw<<<(n2 + 255) / 256, 256, 0, stream>>>(w.FL[2], out + n0 + n1, 16, 16);
  k_nchw<<<(n3 + 255) / 256, 256, 0, stream>>>(w.FL[3], out + n0 + n1 + n2, 8, 8);
}
